// AAConv_24953759989988
// MI455X (gfx1250) — matrix operand order repaired and checked by reading, not compiled, not run on hardware
//
#include <hip/hip_runtime.h>

// ---------------------------------------------------------------------------
// Types for CDNA5 WMMA (wave32, 16x16x32 bf16, f32 accumulate)
// ---------------------------------------------------------------------------
typedef __bf16 v16bf __attribute__((ext_vector_type(16)));
typedef __bf16 v8bf  __attribute__((ext_vector_type(8)));
typedef float  v8f   __attribute__((ext_vector_type(8)));

// Problem constants
#define BB   16
#define CC   256
#define HW   1024     // 32*32
#define DK   128
#define DV   128
#define NH   8
#define DKH  16
#define DVH  16
#define QKVO 384      // 2*DK + DV
#define CONVO 128     // CO - DV

__device__ __forceinline__ v8f wmma_bf16(v16bf a, v16bf b, v8f c) {
  return __builtin_amdgcn_wmma_f32_16x16x32_bf16(false, a, false, b, (short)0, c, false, false);
}

// A-fragment (16x32, bf16) from row-major [M][K] source.
// This code's contiguous k order: lane L holds row M=(L&15); i in 0..15 -> K=(L<16?0:16)+i
// `row` must point at element [m][kbase] for this lane's m.
__device__ __forceinline__ v16bf load_a_frag(const __bf16* row, int lane) {
  int off = (lane < 16) ? 0 : 16;
  v8bf lo = *(const v8bf*)(row + off);
  v8bf hi = *(const v8bf*)(row + off + 8);
  v16bf r;
#pragma unroll
  for (int i = 0; i < 8; ++i) { r[i] = lo[i]; r[8 + i] = hi[i]; }
  return r;
}

__device__ __forceinline__ v16bf zero_frag() {
  v16bf r;
#pragma unroll
  for (int i = 0; i < 16; ++i) r[i] = (__bf16)0.0f;
  return r;
}

// ---------------------------------------------------------------------------
// Kernel 1: weight pack/convert to bf16
//   conv_w HWIO (3,3,256,128) -> Wc[tap][o][c]  (9*128*256)
//   qkv_w  (384,256)          -> Wq[o][c]
//   proj_w (128,128)          -> Wp[o][c]
// ---------------------------------------------------------------------------
__global__ void pack_weights_kernel(const float* __restrict__ conv_w,
                                    const float* __restrict__ qkv_w,
                                    const float* __restrict__ proj_w,
                                    __bf16* __restrict__ Wc,
                                    __bf16* __restrict__ Wq,
                                    __bf16* __restrict__ Wp) {
  int i = blockIdx.x * 256 + threadIdx.x;
  if (i < 9 * 128 * 256) {
    int c = i & 255;
    int o = (i >> 8) & 127;
    int t = i >> 15;
    Wc[i] = (__bf16)conv_w[((size_t)t * 256 + c) * 128 + o];
  }
  if (i < QKVO * 256) Wq[i] = (__bf16)qkv_w[i];
  if (i < 128 * 128)  Wp[i] = (__bf16)proj_w[i];
}

// ---------------------------------------------------------------------------
// Kernel 2: x (B,C,HW) f32  ->  XT (B,HW,C) bf16   (LDS tiled transpose)
// ---------------------------------------------------------------------------
__global__ void transpose_x_kernel(const float* __restrict__ x, __bf16* __restrict__ XT) {
  __shared__ float tile[32][33];
  int b   = blockIdx.z;
  int hw0 = blockIdx.x * 32;
  int c0  = blockIdx.y * 32;
  int tx = threadIdx.x, ty = threadIdx.y;     // block (32,8)
#pragma unroll
  for (int i = 0; i < 4; ++i) {
    int c = c0 + ty + i * 8;
    tile[ty + i * 8][tx] = x[((size_t)b * CC + c) * HW + hw0 + tx];
  }
  __syncthreads();
#pragma unroll
  for (int i = 0; i < 4; ++i) {
    int hw = hw0 + ty + i * 8;
    XT[((size_t)b * HW + hw) * CC + c0 + tx] = (__bf16)tile[tx][ty + i * 8];
  }
}

// ---------------------------------------------------------------------------
// Kernel 3: QKV 1x1-conv GEMM:  kqv[o][hw] = sum_c Wq[o][c] * XT[hw][c]
// Each wave: 16(o) x 64(hw) tile, K=256 in steps of 32.
// Epilogue: +bias, split k/q/v (tf.split order k,q,v), q *= 0.25,
// write Q[b,h,hw,dkh], K[b,h,key,dkh], Vt[b,h,dvh,key] (bf16).
// ---------------------------------------------------------------------------
__global__ void qkv_gemm_kernel(const __bf16* __restrict__ Wq,
                                const __bf16* __restrict__ XT,
                                const float*  __restrict__ qkv_b,
                                __bf16* __restrict__ Qst,
                                __bf16* __restrict__ Kst,
                                __bf16* __restrict__ Vt) {
  int wave = threadIdx.x >> 5;
  int lane = threadIdx.x & 31;
  int job  = blockIdx.x * 4 + wave;       // B * 24 * 16 jobs
  int ng = job % 16;
  int ot = (job / 16) % 24;
  int b  = job / (24 * 16);
  int obase = ot * 16, nbase = ng * 64;
  int orow = obase + (lane & 15);
  int koff = (lane & 16);

  v8f acc[4] = {{}, {}, {}, {}};
  const __bf16* arow = Wq + (size_t)orow * CC;
  for (int cb = 0; cb < CC; cb += 32) {
    v16bf a = load_a_frag(arow + cb, lane);
#pragma unroll
    for (int j = 0; j < 4; ++j) {
      int hw = nbase + j * 16 + (lane & 15);
      v16bf bf = *(const v16bf*)(XT + ((size_t)b * HW + hw) * CC + cb + koff);
      acc[j] = wmma_bf16(a, bf, acc[j]);
    }
  }
#pragma unroll
  for (int j = 0; j < 4; ++j) {
    int hw = nbase + j * 16 + (lane & 15);
#pragma unroll
    for (int r = 0; r < 8; ++r) {
      int o = obase + r + ((lane & 16) ? 8 : 0);
      float v = acc[j][r] + qkv_b[o];
      if (o < DK) {                        // k
        int h = o >> 4, d = o & 15;
        Kst[(((size_t)b * NH + h) * HW + hw) * DKH + d] = (__bf16)v;
      } else if (o < 2 * DK) {             // q (scaled by DKH^-0.5)
        int oq = o - DK; int h = oq >> 4, d = oq & 15;
        Qst[(((size_t)b * NH + h) * HW + hw) * DKH + d] = (__bf16)(v * 0.25f);
      } else {                             // v -> transposed (dvh-major)
        int ov = o - 2 * DK; int h = ov >> 4, d = ov & 15;
        Vt[(((size_t)b * NH + h) * DVH + d) * HW + hw] = (__bf16)v;
      }
    }
  }
}

// ---------------------------------------------------------------------------
// Kernel 4: 3x3 SAME conv as 9 shifted implicit GEMMs over XT.
// out[b][o][hw] = sum_{t,c} Wc[t][o][c] * XT[b][shift_t(hw)][c] + conv_b[o]
// Writes output channels 0..127 (f32).
// ---------------------------------------------------------------------------
__global__ void conv_gemm_kernel(const __bf16* __restrict__ Wc,
                                 const __bf16* __restrict__ XT,
                                 const float*  __restrict__ conv_b,
                                 float* __restrict__ out) {
  int wave = threadIdx.x >> 5;
  int lane = threadIdx.x & 31;
  int job  = blockIdx.x * 4 + wave;       // B * 8 * 16 jobs
  int ng = job & 15;
  int ot = (job >> 4) & 7;
  int b  = job >> 7;
  int obase = ot * 16, nbase = ng * 64;
  int orow = obase + (lane & 15);
  int koff = (lane & 16);

  v8f acc[4] = {{}, {}, {}, {}};
  for (int t = 0; t < 9; ++t) {
    int dy = t / 3 - 1, dx = t % 3 - 1;
    int  hwp[4];
    bool valid[4];
#pragma unroll
    for (int j = 0; j < 4; ++j) {
      int hw = nbase + j * 16 + (lane & 15);
      int y = (hw >> 5) + dy, x = (hw & 31) + dx;
      valid[j] = (y >= 0) && (y < 32) && (x >= 0) && (x < 32);
      hwp[j]   = y * 32 + x;
    }
    const __bf16* arow = Wc + ((size_t)t * CONVO + orow) * CC;
    for (int cb = 0; cb < CC; cb += 32) {
      v16bf a = load_a_frag(arow + cb, lane);
#pragma unroll
      for (int j = 0; j < 4; ++j) {
        v16bf bf = valid[j]
            ? *(const v16bf*)(XT + ((size_t)b * HW + hwp[j]) * CC + cb + koff)
            : zero_frag();
        acc[j] = wmma_bf16(a, bf, acc[j]);
      }
    }
  }
#pragma unroll
  for (int j = 0; j < 4; ++j) {
    int hw = nbase + j * 16 + (lane & 15);
#pragma unroll
    for (int r = 0; r < 8; ++r) {
      int o = obase + r + ((lane & 16) ? 8 : 0);
      out[((size_t)b * 256 + o) * HW + hw] = acc[j][r] + conv_b[o];
    }
  }
}

// ---------------------------------------------------------------------------
// Kernel 5: flash attention. One wave = 16 q-rows of one (b,h).
// QK^T: A=Q(16x32, dkh padded to 32), B=K^T(32x16). Online softmax.
// P routed through per-wave LDS (D-frag -> A-frag relayout), PV: K=32 natural.
// Output attnT[b][hw][h*16+dvh] bf16 (head-major channels).
// ---------------------------------------------------------------------------
__global__ void attn_flash_kernel(const __bf16* __restrict__ Qst,
                                  const __bf16* __restrict__ Kst,
                                  const __bf16* __restrict__ Vt,
                                  __bf16* __restrict__ attnT) {
  __shared__ __bf16 Plds[4][16][32];
  int wave = threadIdx.x >> 5;
  int lane = threadIdx.x & 31;
  int job  = blockIdx.x * 4 + wave;       // B * NH * 64 jobs
  int qt = job & 63;
  int h  = (job >> 6) & 7;
  int b  = job >> 9;
  int qb = qt * 16;

  const __bf16* Qbh = Qst + ((size_t)b * NH + h) * HW * DKH;
  const __bf16* Kbh = Kst + ((size_t)b * NH + h) * HW * DKH;
  const __bf16* Vbh = Vt  + ((size_t)b * NH + h) * DVH * HW;

  // Q A-fragment, dkh contraction zero-padded 16 -> 32
  v16bf qf = zero_frag();
  {
    int row = qb + (lane & 15);
    if (lane < 16) qf = *(const v16bf*)(Qbh + (size_t)row * DKH);
  }

  v8f   acc  = {};
  float mrow[8], lrow[8];
#pragma unroll
  for (int r = 0; r < 8; ++r) { mrow[r] = -3.0e38f; lrow[r] = 0.0f; }

  for (int kt = 0; kt < HW; kt += 32) {
    // K^T B-fragments (kk 16..31 are padding -> zero lanes 16..31)
    v16bf kf0 = zero_frag(), kf1 = zero_frag();
    if (lane < 16) {
      kf0 = *(const v16bf*)(Kbh + (size_t)(kt + lane) * DKH);
      kf1 = *(const v16bf*)(Kbh + (size_t)(kt + 16 + lane) * DKH);
    }
    v8f z = {};
    v8f l0 = wmma_bf16(qf, kf0, z);
    v8f l1 = wmma_bf16(qf, kf1, z);

#pragma unroll
    for (int r = 0; r < 8; ++r) {
      float t = fmaxf(l0[r], l1[r]);
#pragma unroll
      for (int off = 1; off < 16; off <<= 1) t = fmaxf(t, __shfl_xor(t, off));
      float nm   = fmaxf(mrow[r], t);
      float corr = __expf(mrow[r] - nm);
      float p0   = __expf(l0[r] - nm);
      float p1   = __expf(l1[r] - nm);
      float s = p0 + p1;
#pragma unroll
      for (int off = 1; off < 16; off <<= 1) s += __shfl_xor(s, off);
      lrow[r] = lrow[r] * corr + s;
      acc[r] *= corr;
      mrow[r] = nm;
      int m = r + ((lane & 16) ? 8 : 0);
      Plds[wave][m][lane & 15]        = (__bf16)p0;
      Plds[wave][m][16 + (lane & 15)] = (__bf16)p1;
    }
    // LDS is in-order within a wave: loads below see the stores above.
    v16bf pf;
    {
      const __bf16* prow = &Plds[wave][lane & 15][0];
      int off = (lane < 16) ? 0 : 16;
      v8bf lo = *(const v8bf*)(prow + off);
      v8bf hi = *(const v8bf*)(prow + off + 8);
#pragma unroll
      for (int i = 0; i < 8; ++i) { pf[i] = lo[i]; pf[8 + i] = hi[i]; }
    }
    v16bf vf = *(const v16bf*)(Vbh + (size_t)(lane & 15) * HW + kt + (lane & 16));
    acc = wmma_bf16(pf, vf, acc);
  }

#pragma unroll
  for (int r = 0; r < 8; ++r) {
    int m = r + ((lane & 16) ? 8 : 0);
    float o = acc[r] / lrow[r];
    attnT[((size_t)b * HW + qb + m) * DV + h * DVH + (lane & 15)] = (__bf16)o;
  }
}

// ---------------------------------------------------------------------------
// Kernel 6: output projection GEMM: out[128+o][hw] = sum_c Wp[o][c]*attnT[hw][c] + b
// ---------------------------------------------------------------------------
__global__ void proj_gemm_kernel(const __bf16* __restrict__ Wp,
                                 const __bf16* __restrict__ attnT,
                                 const float*  __restrict__ proj_b,
                                 float* __restrict__ out) {
  int wave = threadIdx.x >> 5;
  int lane = threadIdx.x & 31;
  int job  = blockIdx.x * 4 + wave;       // B * 8 * 16 jobs
  int ng = job & 15;
  int ot = (job >> 4) & 7;
  int b  = job >> 7;
  int obase = ot * 16, nbase = ng * 64;
  int orow = obase + (lane & 15);
  int koff = (lane & 16);

  v8f acc[4] = {{}, {}, {}, {}};
  const __bf16* arow = Wp + (size_t)orow * DV;
  for (int cb = 0; cb < DV; cb += 32) {
    v16bf a = load_a_frag(arow + cb, lane);
#pragma unroll
    for (int j = 0; j < 4; ++j) {
      int hw = nbase + j * 16 + (lane & 15);
      v16bf bf = *(const v16bf*)(attnT + ((size_t)b * HW + hw) * DV + cb + koff);
      acc[j] = wmma_bf16(a, bf, acc[j]);
    }
  }
#pragma unroll
  for (int j = 0; j < 4; ++j) {
    int hw = nbase + j * 16 + (lane & 15);
#pragma unroll
    for (int r = 0; r < 8; ++r) {
      int o = obase + r + ((lane & 16) ? 8 : 0);
      out[((size_t)b * 256 + 128 + o) * HW + hw] = acc[j][r] + proj_b[o];
    }
  }
}

// ---------------------------------------------------------------------------
// Host launcher
// ---------------------------------------------------------------------------
extern "C" void kernel_launch(void* const* d_in, const int* in_sizes, int n_in,
                              void* d_out, int out_size, void* d_ws, size_t ws_size,
                              hipStream_t stream) {
  const float* x      = (const float*)d_in[0];
  const float* conv_w = (const float*)d_in[1];
  const float* conv_b = (const float*)d_in[2];
  const float* qkv_w  = (const float*)d_in[3];
  const float* qkv_b  = (const float*)d_in[4];
  const float* proj_w = (const float*)d_in[5];
  const float* proj_b = (const float*)d_in[6];
  float* out = (float*)d_out;

  // carve workspace (256B aligned regions)
  size_t off = 0;
  auto carve = [&](size_t bytes) -> void* {
    void* p = (char*)d_ws + off;
    off += (bytes + 255) & ~(size_t)255;
    return p;
  };
  __bf16* XT    = (__bf16*)carve((size_t)BB * HW * CC * 2);        // 8 MB
  __bf16* Wc    = (__bf16*)carve((size_t)9 * CONVO * CC * 2);      // 0.6 MB
  __bf16* Wq    = (__bf16*)carve((size_t)QKVO * CC * 2);           // 0.2 MB
  __bf16* Wp    = (__bf16*)carve((size_t)DV * DV * 2);             // 32 KB
  __bf16* Qst   = (__bf16*)carve((size_t)BB * NH * HW * DKH * 2);  // 4 MB
  __bf16* Kst   = (__bf16*)carve((size_t)BB * NH * HW * DKH * 2);  // 4 MB
  __bf16* Vt    = (__bf16*)carve((size_t)BB * NH * DVH * HW * 2);  // 4 MB
  __bf16* attnT = (__bf16*)carve((size_t)BB * HW * DV * 2);        // 4 MB

  // 1) pack weights
  pack_weights_kernel<<<dim3((9 * 128 * 256 + 255) / 256), dim3(256), 0, stream>>>(
      conv_w, qkv_w, proj_w, Wc, Wq, Wp);
  // 2) transpose+convert x
  transpose_x_kernel<<<dim3(HW / 32, CC / 32, BB), dim3(32, 8), 0, stream>>>(x, XT);
  // 3) QKV GEMM (+split/bias/scale)
  qkv_gemm_kernel<<<dim3(BB * 24 * 16 / 4), dim3(128), 0, stream>>>(Wq, XT, qkv_b, Qst, Kst, Vt);
  // 4) conv branch (channels 0..127)
  conv_gemm_kernel<<<dim3(BB * 8 * 16 / 4), dim3(128), 0, stream>>>(Wc, XT, conv_b, out);
  // 5) flash attention
  attn_flash_kernel<<<dim3(BB * NH * 64 / 4), dim3(128), 0, stream>>>(Qst, Kst, Vt, attnT);
  // 6) projection (channels 128..255)
  proj_gemm_kernel<<<dim3(BB * 8 * 16 / 4), dim3(128), 0, stream>>>(Wp, attnT, proj_b, out);
}
